// Mamba_51745765982349
// MI455X (gfx1250) — matrix operand order repaired and checked by reading, not compiled, not run on hardware
//
#include <hip/hip_runtime.h>
#include <hip/hip_bf16.h>

// ---------------- problem constants ----------------
#define D_MODEL   256
#define D_STATE   16
#define D_CONV    4
#define D_INNER   512
#define DT_RANK   16
#define BATCH     4
#define SEQLEN    2048
#define M_ROWS    (BATCH * SEQLEN)   // 8192

// ---------------- vector types ----------------
typedef __attribute__((ext_vector_type(16))) __bf16 v16bf;
typedef __attribute__((ext_vector_type(8)))  __bf16 v8bf;
typedef __attribute__((ext_vector_type(8)))  float  v8f;
typedef unsigned int u32x4 __attribute__((ext_vector_type(4)));
typedef int          i32x4 __attribute__((ext_vector_type(4)));
typedef int          i32x8 __attribute__((ext_vector_type(8)));

#if __has_builtin(__builtin_amdgcn_tensor_load_to_lds) && \
    __has_builtin(__builtin_amdgcn_s_wait_tensorcnt)
#define USE_TDM 1
#else
#define USE_TDM 0
#endif

__device__ __forceinline__ unsigned short f2bf(float f) {
  unsigned u = __builtin_bit_cast(unsigned, f);
  u += 0x7FFFu + ((u >> 16) & 1u);          // round-to-nearest-even
  return (unsigned short)(u >> 16);
}

__device__ __forceinline__ v16bf cat16(v8bf lo, v8bf hi) {
  v16bf r;
#pragma unroll
  for (int i = 0; i < 8; ++i) { r[i] = lo[i]; r[i + 8] = hi[i]; }
  return r;
}

// =====================================================================
// one-time f32 -> bf16 conversion (memory-bound elementwise pass)
// =====================================================================
__global__ __launch_bounds__(256)
void f32_to_bf16_kernel(const float* __restrict__ src,
                        unsigned short* __restrict__ dst, int n) {
  int i = (blockIdx.x * 256 + threadIdx.x) * 4;
  if (i >= n) return;
  float4 v = *(const float4*)(src + i);
  unsigned short o[4] = {f2bf(v.x), f2bf(v.y), f2bf(v.z), f2bf(v.w)};
  *(uint2*)(dst + i) = *(const uint2*)o;
}

// =====================================================================
// TDM 2-D tile load: global bf16 (row pitch = strideElems) -> LDS with
// hardware padding: 64B row (16 DWORDs) + 16B pad (4 DWORDs) = 80B pitch.
// Rows beyond rowsValid read as zero (hardware OOB behavior).
// =====================================================================
#if USE_TDM
__device__ __forceinline__ void tdm_load_2d(unsigned ldsAddr, const void* gptr,
                                            int rowsTile, int rowsValid,
                                            int strideElems) {
  unsigned long long ga = (unsigned long long)(size_t)gptr;
  u32x4 g0;
  g0[0] = 1u;                                            // count=1, user mode
  g0[1] = ldsAddr;                                       // lds_addr (bytes)
  g0[2] = (unsigned)(ga & 0xFFFFFFFFu);                  // global_addr lo
  g0[3] = (unsigned)((ga >> 32) & 0x01FFFFFFu) | (2u << 30);  // hi + type=2

  i32x8 g1;
  // data_size=1 (2B) @16 | pad_enable @20 | pad_interval=3 (16 DW) @22 |
  // pad_amount=3 (4 DW) @25 ; workgroup_mask=0
  g1[0] = (1 << 16) | (1 << 20) | (3 << 22) | (3 << 25);
  const unsigned td0 = 32u;                              // tensor_dim0 (tile always valid in K)
  const unsigned td1 = (unsigned)rowsValid;              // tensor_dim1 (valid rows from tile start)
  g1[1] = (int)((td0 & 0xFFFFu) << 16);                  // [47:32] bar addr=0, [63:48] td0.lo
  g1[2] = (int)(((td0 >> 16) & 0xFFFFu) | ((td1 & 0xFFFFu) << 16));
  g1[3] = (int)(((td1 >> 16) & 0xFFFFu) | (32u << 16));  // tile_dim0 = 32 elems
  g1[4] = (int)((unsigned)rowsTile & 0xFFFFu);           // tile_dim1 ; tile_dim2=0 (2D)
  unsigned long long s0 = (unsigned long long)strideElems;
  g1[5] = (int)(s0 & 0xFFFFFFFFu);                       // tensor_dim0_stride lo32
  g1[6] = (int)((s0 >> 32) & 0xFFFFu);                   // stride hi16 ; dim1_stride=0
  g1[7] = 0;

  i32x4 z4 = {0, 0, 0, 0};
#if defined(__clang_major__) && (__clang_major__ >= 23)
  i32x8 z8 = {0, 0, 0, 0, 0, 0, 0, 0};
  __builtin_amdgcn_tensor_load_to_lds(g0, g1, z4, z4, z8, 0);
#else
  __builtin_amdgcn_tensor_load_to_lds(g0, g1, z4, z4, 0);
#endif
}
#endif

// =====================================================================
// bf16 WMMA GEMM:  C[M,N] = A[M,K] * W[N,K]^T   (A, W bf16; C f32)
// block = 256 threads (8 waves), tile = 128(M) x 128(N), K-step = 32.
// Waves arranged 4(M) x 2(N); each wave owns 32x64 = 2x4 WMMA fragments.
// TDM double-buffer pipeline: DMA of tile s+1 overlaps WMMA on tile s.
// =====================================================================
#define KSTEP 32
#define LDSL  40   // LDS row pitch in elements (80 B: 64B data + 16B pad)

__global__ __launch_bounds__(256)
void mamba_gemm_bf16_wmma(const unsigned short* __restrict__ A,
                          const unsigned short* __restrict__ W,
                          float* __restrict__ C,
                          int M, int N, int K) {
  __shared__ __align__(16) unsigned short AsBuf[2][128 * LDSL];
  __shared__ __align__(16) unsigned short WsBuf[2][128 * LDSL];

  const int t      = threadIdx.x;
  const int lane   = t & 31;
  const int wave   = t >> 5;
  const int waveM  = wave & 3;          // 0..3 -> M offset 32*waveM
  const int waveN  = wave >> 2;         // 0..1 -> N offset 64*waveN
  const int blockM = blockIdx.x * 128;
  const int blockN = blockIdx.y * 128;

  v8f acc[2][4];
#pragma unroll
  for (int mi = 0; mi < 2; ++mi)
#pragma unroll
    for (int f = 0; f < 4; ++f)
#pragma unroll
      for (int j = 0; j < 8; ++j) acc[mi][f][j] = 0.0f;

  const int nRemain = N - blockN;
  const int nValid  = nRemain < 128 ? nRemain : 128;
  const int steps   = K / KSTEP;

#if USE_TDM
  unsigned asAddr[2] = {(unsigned)(size_t)(const void*)&AsBuf[0][0],
                        (unsigned)(size_t)(const void*)&AsBuf[1][0]};
  unsigned wsAddr[2] = {(unsigned)(size_t)(const void*)&WsBuf[0][0],
                        (unsigned)(size_t)(const void*)&WsBuf[1][0]};
  if (wave == 0) {   // prologue: DMA tile 0 into buffer 0
    tdm_load_2d(asAddr[0], A + (size_t)blockM * K, 128, 128, K);
    tdm_load_2d(wsAddr[0], W + (size_t)blockN * K, 128, nValid, K);
  }
#else
  const int srow = t >> 1, shalf = t & 1;  // 2 threads per row (16 bf16 each)
#endif

  for (int s = 0; s < steps; ++s) {
    const int buf = s & 1;
#if USE_TDM
    if (wave == 0) __builtin_amdgcn_s_wait_tensorcnt(0);  // tile s landed
    __syncthreads();            // tile s visible; prev reads of other buf done
    if (wave == 0 && s + 1 < steps) {   // DMA tile s+1 behind the compute
      const int k1 = (s + 1) * KSTEP;
      tdm_load_2d(asAddr[buf ^ 1], A + (size_t)blockM * K + k1, 128, 128, K);
      tdm_load_2d(wsAddr[buf ^ 1], W + (size_t)blockN * K + k1, 128, nValid, K);
    }
#else
    { // synchronous bf16 staging into buffer 0
      const int k0 = s * KSTEP;
      {
        const unsigned short* src = A + (size_t)(blockM + srow) * K + k0 + shalf * 16;
        unsigned short* dst = &AsBuf[0][srow * LDSL + shalf * 16];
        *(uint4*)dst       = *(const uint4*)src;
        *(uint4*)(dst + 8) = *(const uint4*)(src + 8);
      }
      {
        const int gn = blockN + srow;
        unsigned short* dst = &WsBuf[0][srow * LDSL + shalf * 16];
        if (gn < N) {
          const unsigned short* src = W + (size_t)gn * K + k0 + shalf * 16;
          *(uint4*)dst       = *(const uint4*)src;
          *(uint4*)(dst + 8) = *(const uint4*)(src + 8);
        } else {
          uint4 z = {0, 0, 0, 0};
          *(uint4*)dst = z;
          *(uint4*)(dst + 8) = z;
        }
      }
    }
    __syncthreads();
#endif

    const unsigned short* Asb = &AsBuf[USE_TDM ? buf : 0][0];
    const unsigned short* Wsb = &WsBuf[USE_TDM ? buf : 0][0];

    // ---- A fragments (ISA 16-bit A layout) ----
    const int kbase = (lane >> 4) << 3;                 // 0 or 8
    v16bf afrag[2];
#pragma unroll
    for (int mi = 0; mi < 2; ++mi) {
      const int am = waveM * 32 + mi * 16 + (lane & 15);
      const unsigned short* ap = &Asb[am * LDSL];
      afrag[mi] = cat16(*(const v8bf*)(ap + kbase),
                        *(const v8bf*)(ap + kbase + 16));
    }
    // ---- B fragments + WMMA ----
    const int bk = (lane >> 4) << 3;                    // 0 or 8
#pragma unroll
    for (int f = 0; f < 4; ++f) {
      const int bn = waveN * 64 + f * 16 + (lane & 15);
      const unsigned short* bp = &Wsb[bn * LDSL + bk];
      v16bf bfrag = cat16(*(const v8bf*)(bp),
                          *(const v8bf*)(bp + 16));
#pragma unroll
      for (int mi = 0; mi < 2; ++mi) {
        acc[mi][f] = __builtin_amdgcn_wmma_f32_16x16x32_bf16(
            false, afrag[mi], false, bfrag, (short)0, acc[mi][f], false, false);
      }
    }
#if !USE_TDM
    __syncthreads();
#endif
  }

  // ---- store C: VGPR j <-> row j + (lane>=16)*8, col = lane%16 ----
#pragma unroll
  for (int mi = 0; mi < 2; ++mi) {
    const int mrow0 = blockM + waveM * 32 + mi * 16 + ((lane >> 4) << 3);
#pragma unroll
    for (int f = 0; f < 4; ++f) {
      const int n = blockN + waveN * 64 + f * 16 + (lane & 15);
      if (n < N) {
#pragma unroll
        for (int j = 0; j < 8; ++j)
          C[(size_t)(mrow0 + j) * N + n] = acc[mi][f][j];
      }
    }
  }
}

// =====================================================================
// depthwise causal conv (K=4) + SiLU.  xz is (B, L, 2*Dn) time-major.
// rev=1: operates on the time-flipped sequence. Output u in scan-time order.
// =====================================================================
__global__ __launch_bounds__(256)
void mamba_conv_silu(const float* __restrict__ xz,
                     const float* __restrict__ w,     // (Dn, 1, 4)
                     const float* __restrict__ bias,  // (Dn,)
                     float* __restrict__ u, int rev) {
  int tid = blockIdx.x * blockDim.x + threadIdx.x;
  int d   = tid & (D_INNER - 1);
  int idx = tid >> 9;                  // b*L + t (scan time)
  int t   = idx & (SEQLEN - 1);
  int b   = idx >> 11;

  float acc = bias[d];
#pragma unroll
  for (int k = 0; k < D_CONV; ++k) {
    int tau = t - (D_CONV - 1) + k;
    if (tau >= 0) {
      int p = rev ? (SEQLEN - 1 - tau) : tau;
      acc = fmaf(w[d * D_CONV + k],
                 xz[(size_t)(b * SEQLEN + p) * (2 * D_INNER) + d], acc);
    }
  }
  acc = acc / (1.0f + __expf(-acc));   // SiLU
  u[(size_t)idx * D_INNER + d] = acc;
}

// =====================================================================
// delta = softplus(dt @ dt_w^T + dt_b)
// =====================================================================
__global__ __launch_bounds__(256)
void mamba_delta(const float* __restrict__ xdbl,   // (B*L, 48)
                 const float* __restrict__ dt_w,   // (Dn, 16)
                 const float* __restrict__ dt_b,   // (Dn,)
                 float* __restrict__ delta) {      // (B*L, Dn)
  int tid = blockIdx.x * blockDim.x + threadIdx.x;
  int d   = tid & (D_INNER - 1);
  int idx = tid >> 9;
  const float* r  = xdbl + (size_t)idx * 48;
  const float* wv = dt_w + d * DT_RANK;
  float acc = dt_b[d];
#pragma unroll
  for (int i = 0; i < DT_RANK; ++i) acc = fmaf(wv[i], r[i], acc);
  delta[(size_t)idx * D_INNER + d] = (acc > 20.0f) ? acc : log1pf(__expf(acc));
}

// =====================================================================
// selective scan: one lane per (b, d) channel, 16-state in VGPRs.
// =====================================================================
__global__ __launch_bounds__(256)
void mamba_scan(const float* __restrict__ delta,   // (B*L, Dn) scan-time
                const float* __restrict__ u,       // (B*L, Dn) scan-time
                const float* __restrict__ xdbl,    // (B*L, 48): B at +16, C at +32
                const float* __restrict__ xz,      // (B, L, 2Dn) original time
                const float* __restrict__ A_log,   // (Dn, 16)
                const float* __restrict__ Dp,      // (Dn,)
                float* __restrict__ y,             // (B, L, Dn) original time
                int rev, int accum) {
  int tid = blockIdx.x * blockDim.x + threadIdx.x;
  int d = tid & (D_INNER - 1);
  int b = tid >> 9;

  float Ac[D_STATE];
#pragma unroll
  for (int n = 0; n < D_STATE; ++n) Ac[n] = -__expf(A_log[d * D_STATE + n]);
  const float Dv = Dp[d];

  float h[D_STATE];
#pragma unroll
  for (int n = 0; n < D_STATE; ++n) h[n] = 0.0f;

  for (int t = 0; t < SEQLEN; ++t) {
    const int idx = b * SEQLEN + t;
    const float dl = delta[(size_t)idx * D_INNER + d];
    const float uu = u[(size_t)idx * D_INNER + d];
    const int p = rev ? (SEQLEN - 1 - t) : t;
    const float z = xz[(size_t)(b * SEQLEN + p) * (2 * D_INNER) + D_INNER + d];

    const float4* Bp = (const float4*)(xdbl + (size_t)idx * 48 + 16);
    const float4* Cp = (const float4*)(xdbl + (size_t)idx * 48 + 32);

    const float du = dl * uu;
    float acc = 0.0f;
#pragma unroll
    for (int q = 0; q < 4; ++q) {
      const float4 Bv = Bp[q];
      const float4 Cv = Cp[q];
      const int n = q * 4;
      float e0 = __expf(dl * Ac[n + 0]);
      float e1 = __expf(dl * Ac[n + 1]);
      float e2 = __expf(dl * Ac[n + 2]);
      float e3 = __expf(dl * Ac[n + 3]);
      h[n + 0] = fmaf(e0, h[n + 0], du * Bv.x); acc = fmaf(h[n + 0], Cv.x, acc);
      h[n + 1] = fmaf(e1, h[n + 1], du * Bv.y); acc = fmaf(h[n + 1], Cv.y, acc);
      h[n + 2] = fmaf(e2, h[n + 2], du * Bv.z); acc = fmaf(h[n + 2], Cv.z, acc);
      h[n + 3] = fmaf(e3, h[n + 3], du * Bv.w); acc = fmaf(h[n + 3], Cv.w, acc);
    }
    const float sig = 1.0f / (1.0f + __expf(-z));
    const float val = 0.5f * (acc + uu * Dv) * (z * sig);
    const size_t o = (size_t)(b * SEQLEN + p) * D_INNER + d;
    if (accum) y[o] += val; else y[o] = val;
  }
}

// =====================================================================
// host side
// =====================================================================
struct StreamParams {
  const float *in_w, *conv_w, *conv_b, *x_w, *dt_w, *dt_b, *D;
  const float *conv_w_b, *conv_b_b, *x_w_b, *dt_w_b, *dt_b_b, *D_b, *out_w;
};

extern "C" void kernel_launch(void* const* d_in, const int* in_sizes, int n_in,
                              void* d_out, int out_size, void* d_ws, size_t ws_size,
                              hipStream_t stream) {
  const float* hs[2]   = {(const float*)d_in[0], (const float*)d_in[1]};
  const float* A_log   = (const float*)d_in[2];
  const float* A_b_log = (const float*)d_in[3];

  StreamParams sp[2];
  for (int s = 0; s < 2; ++s) {
    int o = 4 + s * 14;
    sp[s].in_w     = (const float*)d_in[o + 0];
    sp[s].conv_w   = (const float*)d_in[o + 1];
    sp[s].conv_b   = (const float*)d_in[o + 2];
    sp[s].x_w      = (const float*)d_in[o + 3];
    sp[s].dt_w     = (const float*)d_in[o + 4];
    sp[s].dt_b     = (const float*)d_in[o + 5];
    sp[s].D        = (const float*)d_in[o + 6];
    sp[s].conv_w_b = (const float*)d_in[o + 7];
    sp[s].conv_b_b = (const float*)d_in[o + 8];
    sp[s].x_w_b    = (const float*)d_in[o + 9];
    sp[s].dt_w_b   = (const float*)d_in[o + 10];
    sp[s].dt_b_b   = (const float*)d_in[o + 11];
    sp[s].D_b      = (const float*)d_in[o + 12];
    sp[s].out_w    = (const float*)d_in[o + 13];
  }

  // ---- workspace layout (bytes), reused across sequential streams/dirs ----
  char* wsb = (char*)d_ws;
  float* xz   = (float*)wsb; wsb += (size_t)M_ROWS * 2 * D_INNER * 4;  // 32 MB
  float* ub   = (float*)wsb; wsb += (size_t)M_ROWS * D_INNER * 4;      // 16 MB
  float* xdbl = (float*)wsb; wsb += (size_t)M_ROWS * 48 * 4;           // 1.5 MB
  float* dl   = (float*)wsb; wsb += (size_t)M_ROWS * D_INNER * 4;      // 16 MB
  float* yc   = (float*)wsb; wsb += (size_t)M_ROWS * D_INNER * 4;      // 16 MB
  unsigned short* Ab = (unsigned short*)wsb; wsb += (size_t)M_ROWS * D_INNER * 2;   // 8 MB
  unsigned short* Wb = (unsigned short*)wsb; wsb += (size_t)(2 * D_INNER) * D_MODEL * 2; // 0.5 MB

  const dim3 blk(256);
  const int elemGrd = (M_ROWS * D_INNER) / 256;
  const int scanGrd = (BATCH * D_INNER) / 256;

  auto cvt = [&](const float* s, unsigned short* d, int n) {
    f32_to_bf16_kernel<<<(n / 4 + 255) / 256, blk, 0, stream>>>(s, d, n);
  };

  for (int s = 0; s < 2; ++s) {
    const StreamParams& P = sp[s];
    float* out_seg = (float*)d_out + (size_t)s * M_ROWS * D_MODEL;

    // in_proj: xz(B,L,1024) = h(B,L,256) @ in_w(1024,256)^T
    cvt(hs[s], Ab, M_ROWS * D_MODEL);
    cvt(P.in_w, Wb, (2 * D_INNER) * D_MODEL);
    mamba_gemm_bf16_wmma<<<dim3(M_ROWS / 128, (2 * D_INNER) / 128), blk, 0, stream>>>(
        Ab, Wb, xz, M_ROWS, 2 * D_INNER, D_MODEL);

    // ---- forward branch ----
    mamba_conv_silu<<<elemGrd, blk, 0, stream>>>(xz, P.conv_w, P.conv_b, ub, 0);
    cvt(ub, Ab, M_ROWS * D_INNER);
    cvt(P.x_w, Wb, (DT_RANK + 2 * D_STATE) * D_INNER);
    mamba_gemm_bf16_wmma<<<dim3(M_ROWS / 128, 1), blk, 0, stream>>>(
        Ab, Wb, xdbl, M_ROWS, DT_RANK + 2 * D_STATE, D_INNER);
    mamba_delta<<<elemGrd, blk, 0, stream>>>(xdbl, P.dt_w, P.dt_b, dl);
    mamba_scan<<<scanGrd, blk, 0, stream>>>(dl, ub, xdbl, xz, A_log, P.D, yc, 0, 0);

    // ---- reverse branch ----
    mamba_conv_silu<<<elemGrd, blk, 0, stream>>>(xz, P.conv_w_b, P.conv_b_b, ub, 1);
    cvt(ub, Ab, M_ROWS * D_INNER);
    cvt(P.x_w_b, Wb, (DT_RANK + 2 * D_STATE) * D_INNER);
    mamba_gemm_bf16_wmma<<<dim3(M_ROWS / 128, 1), blk, 0, stream>>>(
        Ab, Wb, xdbl, M_ROWS, DT_RANK + 2 * D_STATE, D_INNER);
    mamba_delta<<<elemGrd, blk, 0, stream>>>(xdbl, P.dt_w_b, P.dt_b_b, dl);
    mamba_scan<<<scanGrd, blk, 0, stream>>>(dl, ub, xdbl, xz, A_b_log, P.D_b, yc, 1, 1);

    // out_proj: out(B,L,256) = yc(B,L,512) @ out_w(256,512)^T
    cvt(yc, Ab, M_ROWS * D_INNER);
    cvt(P.out_w, Wb, D_MODEL * D_INNER);
    mamba_gemm_bf16_wmma<<<dim3(M_ROWS / 128, D_MODEL / 128), blk, 0, stream>>>(
        Ab, Wb, out_seg, M_ROWS, D_MODEL, D_INNER);
  }
}
